// MessagePassingLayer_47528108097775
// MI455X (gfx1250) — matrix operand order repaired and checked by reading, not compiled, not run on hardware
//
#include <hip/hip_runtime.h>

typedef __attribute__((ext_vector_type(16))) _Float16 v16h;
typedef __attribute__((ext_vector_type(8)))  _Float16 v8h;
typedef __attribute__((ext_vector_type(8)))  float    v8f;

// ---------------- problem constants ----------------
static constexpr int Nn = 20000;
static constexpr int Fd = 128;
static constexpr int Ee = 320000;
static constexpr int FE = 16;
static constexpr int Hd = 128;
static constexpr int NT = 2;
static constexpr int NP = 4;

static constexpr int SLOTS_E = 320512;   // multiple of 128, >= E + 4*128
static constexpr int SLOTS_N = 20352;    // multiple of 128, >= N + 2*128
static constexpr int WG_E = SLOTS_E / 128;   // 2504 workgroups (8 waves * 16 edges)
static constexpr int WG_N = SLOTS_N / 128;   // 159

// ---------------- workspace layout (bytes) ----------------
static constexpr size_t OFF_AGG   = 0;                               // N*H f32
static constexpr size_t OFF_PERME = OFF_AGG + (size_t)Nn * Hd * 4;   // 10,240,000
static constexpr size_t OFF_PERMN = OFF_PERME + (size_t)SLOTS_E * 4;
static constexpr size_t OFF_CTR   = OFF_PERMN + (size_t)SLOTS_N * 4; // 32 ints
static constexpr size_t OFF_W1E   = (OFF_CTR + 2048) & ~(size_t)1023;
static constexpr size_t W1E_BYTES = (size_t)NP * 9 * 8 * 1024;       // 294912
static constexpr size_t OFF_W2E   = OFF_W1E + W1E_BYTES;
static constexpr size_t W2E_BYTES = (size_t)NP * 4 * 8 * 1024;       // 131072
static constexpr size_t OFF_W1N   = OFF_W2E + W2E_BYTES;
static constexpr size_t W1N_BYTES = (size_t)NT * 8 * 8 * 1024;       // 131072
static constexpr size_t OFF_W2N   = OFF_W1N + W1N_BYTES;
static constexpr size_t W2N_BYTES = (size_t)NT * 4 * 8 * 1024;       // 65536

// ctr indices: [0..3] cntE, [4..7] baseE, [8..11] cursE,
//              [12..13] cntN, [14..15] baseN, [16..17] cursN

// ---------------- small setup kernels ----------------
__global__ void kinit(float* agg, unsigned* permE, unsigned* permN, int* ctr)
{
    const int stride = gridDim.x * blockDim.x;
    const int t0 = blockIdx.x * blockDim.x + threadIdx.x;
    for (int i = t0; i < Nn * Hd; i += stride) agg[i] = 0.f;
    for (int i = t0; i < SLOTS_E; i += stride) permE[i] = 0xFFFFFFFFu;
    for (int i = t0; i < SLOTS_N; i += stride) permN[i] = 0xFFFFFFFFu;
    for (int i = t0; i < 32;      i += stride) ctr[i] = 0;
}

__global__ void khist(const long long* __restrict__ eidx,
                      const long long* __restrict__ ntype, int* ctr)
{
    const int stride = gridDim.x * blockDim.x;
    const int t0 = blockIdx.x * blockDim.x + threadIdx.x;
    for (int i = t0; i < Ee; i += stride) {
        int r = (int)eidx[i], c = (int)eidx[Ee + i];
        int p = (int)ntype[r] * NT + (int)ntype[c];
        atomicAdd(&ctr[p], 1);
    }
    for (int i = t0; i < Nn; i += stride)
        atomicAdd(&ctr[12 + (int)ntype[i]], 1);
}

__global__ void kprefix(int* ctr)
{
    if (threadIdx.x == 0 && blockIdx.x == 0) {
        int b = 0;
        for (int p = 0; p < NP; ++p) {
            ctr[4 + p] = b; ctr[8 + p] = b;
            b = (b + ctr[p] + 127) & ~127;          // pad buckets to 128 slots
        }
        int bn = 0;
        for (int t = 0; t < NT; ++t) {
            ctr[14 + t] = bn; ctr[16 + t] = bn;
            bn = (bn + ctr[12 + t] + 127) & ~127;
        }
    }
}

__global__ void kscatter(const long long* __restrict__ eidx,
                         const long long* __restrict__ ntype, int* ctr,
                         unsigned* permE, unsigned* permN)
{
    const int stride = gridDim.x * blockDim.x;
    const int t0 = blockIdx.x * blockDim.x + threadIdx.x;
    for (int i = t0; i < Ee; i += stride) {
        int r = (int)eidx[i], c = (int)eidx[Ee + i];
        int p = (int)ntype[r] * NT + (int)ntype[c];
        int pos = atomicAdd(&ctr[8 + p], 1);
        permE[pos] = (unsigned)i;
    }
    for (int i = t0; i < Nn; i += stride) {
        int pos = atomicAdd(&ctr[16 + (int)ntype[i]], 1);
        permN[pos] = (unsigned)i;
    }
}

// Pack fp32 weights into f16 WMMA B-fragments.
// B-fragment (16x16x32 f16): lane holds column n = lane%16;
// half h (0..15) holds k = 32*c + 8*(lane/16) + h for h<8, 32*c + 16 + 8*(lane/16) + (h-8) for h>=8.
__global__ void kfrag(const float* __restrict__ ew1, const float* __restrict__ ew2,
                      const float* __restrict__ nw1, const float* __restrict__ nw2,
                      _Float16* w1e, _Float16* w2e, _Float16* w1n, _Float16* w2n)
{
    int idx = blockIdx.x * blockDim.x + threadIdx.x;
    if (idx >= 608 * 32) return;
    int fid = idx >> 5, lane = idx & 31;
    int n = lane & 15, gg = lane >> 4;
    const float* src; _Float16* dst; int c, cb, Kv;
    if (fid < 288) {                   // ew1: (4, 272, 128) -> 9 k-chunks
        int p = fid / 72, r = fid % 72; c = r / 8; cb = r % 8;
        src = ew1 + (size_t)p * 272 * 128; Kv = 272; dst = w1e + (size_t)fid * 512;
    } else if (fid < 416) {            // ew2: (4, 128, 128) -> 4 k-chunks
        int f = fid - 288; int p = f / 32, r = f % 32; c = r / 8; cb = r % 8;
        src = ew2 + (size_t)p * 128 * 128; Kv = 128; dst = w2e + (size_t)f * 512;
    } else if (fid < 544) {            // nw1: (2, 256, 128) -> 8 k-chunks
        int f = fid - 416; int t = f / 64, r = f % 64; c = r / 8; cb = r % 8;
        src = nw1 + (size_t)t * 256 * 128; Kv = 256; dst = w1n + (size_t)f * 512;
    } else {                           // nw2: (2, 128, 128) -> 4 k-chunks
        int f = fid - 544; int t = f / 32, r = f % 32; c = r / 8; cb = r % 8;
        src = nw2 + (size_t)t * 128 * 128; Kv = 128; dst = w2n + (size_t)f * 512;
    }
    int col = cb * 16 + n;
#pragma unroll
    for (int h = 0; h < 16; ++h) {
        int k = c * 32 + (h < 8 ? 8 * gg + h : 16 + 8 * gg + (h - 8));
        float v = (k < Kv) ? src[(size_t)k * 128 + col] : 0.f;
        dst[lane * 16 + h] = (_Float16)v;
    }
}

// ---------------- WMMA helpers ----------------
__device__ inline v16h lds16(const _Float16* p)
{
    v8h lo = *(const v8h*)p;
    v8h hi = *(const v8h*)(p + 8);
    return __builtin_shufflevector(lo, hi, 0,1,2,3,4,5,6,7,8,9,10,11,12,13,14,15);
}

// 8 consecutive fp32 from the concatenated edge input [x_row | x_col | e_attr | 0pad]
__device__ inline void runE(float* f, int k, bool valid,
                            const float* __restrict__ x, const float* __restrict__ ea,
                            int row, int col, unsigned e)
{
    const float* src = nullptr;
    if (valid) {
        if (k < 128)      src = x  + (size_t)row * Fd + k;
        else if (k < 256) src = x  + (size_t)col * Fd + (k - 128);
        else if (k < 272) src = ea + (size_t)e   * FE + (k - 256);
    }
    if (src) {
        float4 a = ((const float4*)src)[0];
        float4 b = ((const float4*)src)[1];
        f[0]=a.x; f[1]=a.y; f[2]=a.z; f[3]=a.w;
        f[4]=b.x; f[5]=b.y; f[6]=b.z; f[7]=b.w;
    } else {
#pragma unroll
        for (int i = 0; i < 8; ++i) f[i] = 0.f;
    }
}

__device__ inline void runN(float* f, int k, bool valid,
                            const float* __restrict__ x, const float* __restrict__ agg,
                            int node)
{
    const float* src = nullptr;
    if (valid) {
        if (k < 128) src = x   + (size_t)node * Fd + k;
        else         src = agg + (size_t)node * Hd + (k - 128);
    }
    if (src) {
        float4 a = ((const float4*)src)[0];
        float4 b = ((const float4*)src)[1];
        f[0]=a.x; f[1]=a.y; f[2]=a.z; f[3]=a.w;
        f[4]=b.x; f[5]=b.y; f[6]=b.z; f[7]=b.w;
    } else {
#pragma unroll
        for (int i = 0; i < 8; ++i) f[i] = 0.f;
    }
}

// ---------------- edge MLP + scatter ----------------
// WG = 256 threads = 8 waves; each wave owns one 16-edge tile; bucket padding
// guarantees the whole WG shares one pair id p.
__global__ void __launch_bounds__(256) kedge(
    const float* __restrict__ x, const long long* __restrict__ eidx,
    const float* __restrict__ eattr,
    const float* __restrict__ eb1, const float* __restrict__ eb2,
    const _Float16* __restrict__ w1f, const _Float16* __restrict__ w2f,
    const unsigned* __restrict__ perm, const int* __restrict__ ctr,
    float* __restrict__ agg)
{
    extern __shared__ char smem[];
    _Float16* sW1  = (_Float16*)smem;                 // 73728 B (9*8 frags)
    _Float16* sW2  = (_Float16*)(smem + 73728);       // 32768 B (4*8 frags)
    _Float16* sStg = (_Float16*)(smem + 106496);      // 8 waves * 16x136 f16

    const int tid = threadIdx.x;
    const int wave = tid >> 5, lane = tid & 31;
    const int m = lane & 15, g = lane >> 4;
    const int slot0 = blockIdx.x * 128;

    int p = 0;
    if (slot0 >= ctr[5]) p = 1;
    if (slot0 >= ctr[6]) p = 2;
    if (slot0 >= ctr[7]) p = 3;

    {   // stage this pair's weights into LDS (reused by 8 tiles)
        const float4* s1 = (const float4*)(w1f + (size_t)p * 36864);
        const float4* s2 = (const float4*)(w2f + (size_t)p * 16384);
        float4* d1 = (float4*)sW1;
        float4* d2 = (float4*)sW2;
        for (int i = tid; i < 4608; i += 256) d1[i] = s1[i];
        for (int i = tid; i < 2048; i += 256) d2[i] = s2[i];
    }
    __syncthreads();

    const int slotBase = slot0 + wave * 16;
    const unsigned e = perm[slotBase + m];
    const bool valid = (e != 0xFFFFFFFFu);
    int row = 0, col = 0;
    if (valid) { row = (int)eidx[e]; col = (int)eidx[Ee + e]; }

    // ---- layer 1: (16 x 288) @ (288 x 128) ----
    v8f acc[8] = {};
#pragma unroll
    for (int c = 0; c < 9; ++c) {
        v16h a;
        {
            float f0[8], f1[8];
            runE(f0, c * 32 + 8 * g,      valid, x, eattr, row, col, e);
            runE(f1, c * 32 + 16 + 8 * g, valid, x, eattr, row, col, e);
#pragma unroll
            for (int i = 0; i < 8; ++i) { a[i] = (_Float16)f0[i]; a[8 + i] = (_Float16)f1[i]; }
        }
#pragma unroll
        for (int cb = 0; cb < 8; ++cb) {
            v16h b = lds16(sW1 + ((size_t)(c * 8 + cb) * 32 + lane) * 16);
            acc[cb] = __builtin_amdgcn_wmma_f32_16x16x32_f16(
                false, a, false, b, (short)0, acc[cb], false, false);
        }
    }

    // bias + relu, restage as f16 row-major for layer-2 A fragments
    _Float16* st = sStg + wave * 2176;                // 16 rows * 136 halves
#pragma unroll
    for (int cb = 0; cb < 8; ++cb) {
        float bias = eb1[p * Hd + cb * 16 + m];
#pragma unroll
        for (int v = 0; v < 8; ++v) {
            float t = acc[cb][v] + bias;
            st[(v + 8 * g) * 136 + cb * 16 + m] = (_Float16)(t > 0.f ? t : 0.f);
        }
    }
    __syncthreads();

    // ---- layer 2: (16 x 128) @ (128 x 128) ----
    v8f acc2[8] = {};
#pragma unroll
    for (int c = 0; c < 4; ++c) {
        v8h lo = *(const v8h*)(st + m * 136 + c * 32 + 8 * g);
        v8h hi = *(const v8h*)(st + m * 136 + c * 32 + 16 + 8 * g);
        v16h a = __builtin_shufflevector(lo, hi, 0,1,2,3,4,5,6,7,8,9,10,11,12,13,14,15);
#pragma unroll
        for (int cb = 0; cb < 8; ++cb) {
            v16h b = lds16(sW2 + ((size_t)(c * 8 + cb) * 32 + lane) * 16);
            acc2[cb] = __builtin_amdgcn_wmma_f32_16x16x32_f16(
                false, a, false, b, (short)0, acc2[cb], false, false);
        }
    }

    // bias + relu + atomic segment-sum into agg[row] (L2-resident)
#pragma unroll
    for (int v = 0; v < 8; ++v) {
        unsigned e2 = perm[slotBase + v + 8 * g];
        if (e2 != 0xFFFFFFFFu) {
            int r2 = (int)eidx[e2];
            float* dst = agg + (size_t)r2 * Hd + m;
#pragma unroll
            for (int cb = 0; cb < 8; ++cb) {
                float t = acc2[cb][v] + eb2[p * Hd + cb * 16 + m];
                t = t > 0.f ? t : 0.f;
                atomicAdd(dst + cb * 16, t);
            }
        }
    }
}

// ---------------- node MLP + residual ----------------
__global__ void __launch_bounds__(256) knode(
    const float* __restrict__ x, const float* __restrict__ agg,
    const float* __restrict__ nb1, const float* __restrict__ nb2,
    const _Float16* __restrict__ w1f, const _Float16* __restrict__ w2f,
    const unsigned* __restrict__ perm, const int* __restrict__ ctr,
    float* __restrict__ out)
{
    extern __shared__ char smem[];
    _Float16* sW1  = (_Float16*)smem;                 // 65536 B (8*8 frags)
    _Float16* sW2  = (_Float16*)(smem + 65536);       // 32768 B (4*8 frags)
    _Float16* sStg = (_Float16*)(smem + 98304);

    const int tid = threadIdx.x;
    const int wave = tid >> 5, lane = tid & 31;
    const int m = lane & 15, g = lane >> 4;
    const int slot0 = blockIdx.x * 128;

    const int t = (slot0 >= ctr[15]) ? 1 : 0;

    {
        const float4* s1 = (const float4*)(w1f + (size_t)t * 32768);
        const float4* s2 = (const float4*)(w2f + (size_t)t * 16384);
        float4* d1 = (float4*)sW1;
        float4* d2 = (float4*)sW2;
        for (int i = tid; i < 4096; i += 256) d1[i] = s1[i];
        for (int i = tid; i < 2048; i += 256) d2[i] = s2[i];
    }
    __syncthreads();

    const int slotBase = slot0 + wave * 16;
    const unsigned nd = perm[slotBase + m];
    const bool valid = (nd != 0xFFFFFFFFu);
    const int node = valid ? (int)nd : 0;

    // ---- layer 1: (16 x 256) @ (256 x 128) ----
    v8f acc[8] = {};
#pragma unroll
    for (int c = 0; c < 8; ++c) {
        v16h a;
        {
            float f0[8], f1[8];
            runN(f0, c * 32 + 8 * g,      valid, x, agg, node);
            runN(f1, c * 32 + 16 + 8 * g, valid, x, agg, node);
#pragma unroll
            for (int i = 0; i < 8; ++i) { a[i] = (_Float16)f0[i]; a[8 + i] = (_Float16)f1[i]; }
        }
#pragma unroll
        for (int cb = 0; cb < 8; ++cb) {
            v16h b = lds16(sW1 + ((size_t)(c * 8 + cb) * 32 + lane) * 16);
            acc[cb] = __builtin_amdgcn_wmma_f32_16x16x32_f16(
                false, a, false, b, (short)0, acc[cb], false, false);
        }
    }

    _Float16* st = sStg + wave * 2176;
#pragma unroll
    for (int cb = 0; cb < 8; ++cb) {
        float bias = nb1[t * Hd + cb * 16 + m];
#pragma unroll
        for (int v = 0; v < 8; ++v) {
            float q = acc[cb][v] + bias;
            st[(v + 8 * g) * 136 + cb * 16 + m] = (_Float16)(q > 0.f ? q : 0.f);
        }
    }
    __syncthreads();

    // ---- layer 2 (no relu): (16 x 128) @ (128 x 128) ----
    v8f acc2[8] = {};
#pragma unroll
    for (int c = 0; c < 4; ++c) {
        v8h lo = *(const v8h*)(st + m * 136 + c * 32 + 8 * g);
        v8h hi = *(const v8h*)(st + m * 136 + c * 32 + 16 + 8 * g);
        v16h a = __builtin_shufflevector(lo, hi, 0,1,2,3,4,5,6,7,8,9,10,11,12,13,14,15);
#pragma unroll
        for (int cb = 0; cb < 8; ++cb) {
            v16h b = lds16(sW2 + ((size_t)(c * 8 + cb) * 32 + lane) * 16);
            acc2[cb] = __builtin_amdgcn_wmma_f32_16x16x32_f16(
                false, a, false, b, (short)0, acc2[cb], false, false);
        }
    }

    // bias + residual x0[:, :OUT], scatter to out[node]
#pragma unroll
    for (int v = 0; v < 8; ++v) {
        unsigned n2 = perm[slotBase + v + 8 * g];
        if (n2 != 0xFFFFFFFFu) {
#pragma unroll
            for (int cb = 0; cb < 8; ++cb) {
                size_t o = (size_t)n2 * Fd + cb * 16 + m;
                out[o] = acc2[cb][v] + nb2[t * Hd + cb * 16 + m] + x[o];
            }
        }
    }
}

// ---------------- host launcher ----------------
extern "C" void kernel_launch(void* const* d_in, const int* in_sizes, int n_in,
                              void* d_out, int out_size, void* d_ws, size_t ws_size,
                              hipStream_t stream)
{
    const float*     x     = (const float*)d_in[0];
    const long long* eidx  = (const long long*)d_in[1];
    const float*     eattr = (const float*)d_in[2];
    const long long* ntype = (const long long*)d_in[3];
    const float*     ew1   = (const float*)d_in[4];
    const float*     eb1   = (const float*)d_in[5];
    const float*     ew2   = (const float*)d_in[6];
    const float*     eb2   = (const float*)d_in[7];
    const float*     nw1   = (const float*)d_in[8];
    const float*     nb1   = (const float*)d_in[9];
    const float*     nw2   = (const float*)d_in[10];
    const float*     nb2   = (const float*)d_in[11];
    float* out = (float*)d_out;
    char*  ws  = (char*)d_ws;

    float*     agg   = (float*)(ws + OFF_AGG);
    unsigned*  permE = (unsigned*)(ws + OFF_PERME);
    unsigned*  permN = (unsigned*)(ws + OFF_PERMN);
    int*       ctr   = (int*)(ws + OFF_CTR);
    _Float16*  w1e   = (_Float16*)(ws + OFF_W1E);
    _Float16*  w2e   = (_Float16*)(ws + OFF_W2E);
    _Float16*  w1n   = (_Float16*)(ws + OFF_W1N);
    _Float16*  w2n   = (_Float16*)(ws + OFF_W2N);

    (void)in_sizes; (void)n_in; (void)out_size; (void)ws_size;

    hipFuncSetAttribute((const void*)kedge, hipFuncAttributeMaxDynamicSharedMemorySize, 141312);
    hipFuncSetAttribute((const void*)knode, hipFuncAttributeMaxDynamicSharedMemorySize, 133120);

    kinit   <<<4096, 256, 0, stream>>>(agg, permE, permN, ctr);
    khist   <<<1280, 256, 0, stream>>>(eidx, ntype, ctr);
    kprefix <<<1, 1, 0, stream>>>(ctr);
    kscatter<<<1280, 256, 0, stream>>>(eidx, ntype, ctr, permE, permN);
    kfrag   <<<76, 256, 0, stream>>>(ew1, ew2, nw1, nw2, w1e, w2e, w1n, w2n);
    kedge   <<<WG_E, 256, 141312, stream>>>(x, eidx, eattr, eb1, eb2, w1e, w2e, permE, ctr, agg);
    knode   <<<WG_N, 256, 133120, stream>>>(x, agg, nb1, nb2, w1n, w2n, permN, ctr, out);
}
